// Model_89945205113565
// MI455X (gfx1250) — matrix operand order repaired and checked by reading, not compiled, not run on hardware
//
#include <hip/hip_runtime.h>
#include <hip/hip_bf16.h>
#include <math.h>

// ---------------------------------------------------------------------------
// GAIN (GRU-GAN) forward for MI455X / gfx1250.
//   * All GEMMs in bf16 via v_wmma_f32_16x16x32_bf16 (wave32 WMMA).
//   * Recurrent scan: persistent kernel; Whh B-fragments hoisted into VGPRs
//     for the whole 512-step scan, h A-fragments loaded from LDS once per
//     step and reused across N-tiles -> inner loop is back-to-back WMMA.
//   * Input-projection GEMM: 2 N-tiles per wave sharing one A fragment.
//   * Per-batch top-k mask via bitonic sort in LDS (64KB of 320KB WGP LDS).
// ---------------------------------------------------------------------------

typedef __attribute__((ext_vector_type(16))) __bf16 v16bf;
typedef __attribute__((ext_vector_type(8)))  float  v8f;

union FragBF { v16bf v; uint4 u[2]; };

__device__ __forceinline__ float wave_sum(float v) {
  #pragma unroll
  for (int off = 16; off > 0; off >>= 1) v += __shfl_down(v, off, 32);
  return v;
}

// ---------------------------------------------------------------------------
// 1) Per-batch mask: drop the top-k of mask_rand*obs_mask (k = round(#obs * r))
// ---------------------------------------------------------------------------
__global__ void mask_topk_kernel(const float* __restrict__ obs,
                                 const float* __restrict__ rnd,
                                 const float* __restrict__ ratio,
                                 float* __restrict__ M) {
  const int N = 512 * 32; // 16384 per batch sample
  extern __shared__ float s[];
  __shared__ float wsum[32];
  __shared__ int   s_k;
  __shared__ float s_th;
  const int b = blockIdx.x, tid = threadIdx.x;
  const float* ob = obs + (size_t)b * N;
  const float* rn = rnd + (size_t)b * N;

  float cnt = 0.f;
  for (int i = tid; i < N; i += blockDim.x) {
    float o = ob[i];
    s[i] = rn[i] * o;
    cnt += o;
  }
  cnt = wave_sum(cnt);
  if ((tid & 31) == 0) wsum[tid >> 5] = cnt;
  __syncthreads();
  if (tid < 32) {
    float t = (tid < (int)(blockDim.x >> 5)) ? wsum[tid] : 0.f;
    t = wave_sum(t);
    if (tid == 0) s_k = (int)rintf(t * ratio[b]);
  }
  for (int size = 2; size <= N; size <<= 1) {
    for (int stride = size >> 1; stride > 0; stride >>= 1) {
      __syncthreads();
      for (int p = tid; p < N / 2; p += blockDim.x) {
        int lo = 2 * p - (p & (stride - 1));
        int hi = lo + stride;
        bool up = ((lo & size) == 0);
        float a = s[lo], c = s[hi];
        if ((a > c) == up) { s[lo] = c; s[hi] = a; }
      }
    }
  }
  __syncthreads();
  if (tid == 0) {
    int k = s_k;
    s_th = (k <= 0) ? 3.4e38f : ((k >= N) ? -3.4e38f : s[N - k]);
  }
  __syncthreads();
  const float th = s_th;
  for (int i = tid; i < N; i += blockDim.x) {
    float v = rn[i] * ob[i];
    M[(size_t)b * N + i] = (v > 0.f && v < th) ? 1.f : 0.f;
  }
}

// ---------------------------------------------------------------------------
// 2) fp32 -> bf16 conversion
// ---------------------------------------------------------------------------
__global__ void f32_to_bf16_kernel(const float* __restrict__ in,
                                   __bf16* __restrict__ out, int n) {
  for (int i = blockIdx.x * blockDim.x + threadIdx.x; i < n;
       i += gridDim.x * blockDim.x)
    out[i] = (__bf16)in[i];
}

// ---------------------------------------------------------------------------
// 3) Elementwise builders
// ---------------------------------------------------------------------------
__global__ void build_gen_input_kernel(const float* __restrict__ data,
                                       const float* __restrict__ Z,
                                       const float* __restrict__ M,
                                       float* __restrict__ X,
                                       __bf16* __restrict__ act, int n) {
  int i = blockIdx.x * blockDim.x + threadIdx.x;
  if (i >= n) return;
  float m = M[i];
  float x = m * data[i] + (1.f - m) * Z[i];
  X[i] = x;
  size_t row = (size_t)(i >> 5);
  int c = i & 31;
  act[row * 64 + c]      = (__bf16)x;
  act[row * 64 + 32 + c] = (__bf16)m;
}

__global__ void build_disc_input_kernel(const float* __restrict__ X,
                                        const float* __restrict__ G,
                                        const float* __restrict__ M,
                                        const float* __restrict__ Ht,
                                        __bf16* __restrict__ act, int n) {
  int i = blockIdx.x * blockDim.x + threadIdx.x;
  if (i >= n) return;
  float m  = M[i];
  float xh = X[i] * m + G[i] * (1.f - m);
  float hh = m * Ht[i];
  size_t row = (size_t)(i >> 5);
  int c = i & 31;
  act[row * 64 + c]      = (__bf16)xh;
  act[row * 64 + 32 + c] = (__bf16)hh;
}

// ---------------------------------------------------------------------------
// 4) Input-projection GEMM: C[M,N] = A[M,K](bf16) * W[N,K]^T(bf16) + bias.
//    4 waves/block; each wave computes TWO adjacent 16x16 N-tiles sharing one
//    A fragment (halves A traffic, doubles WMMA per load clause).
// ---------------------------------------------------------------------------
__global__ void gemm_bias_kernel(const __bf16* __restrict__ A,
                                 const __bf16* __restrict__ W,
                                 const float* __restrict__ bias,
                                 float* __restrict__ C, int K, int N) {
  const int lane  = threadIdx.x & 31;
  const int wave  = threadIdx.x >> 5;
  const int tileM = blockIdx.y * 4 + wave;
  const int col0  = blockIdx.x * 32;                    // two N-tiles
  const size_t arow = (size_t)tileM * 16 + (lane & 15);
  const int koffA = (lane >> 4) * 16;
  const int bcol0 = col0 + (lane & 15);
  const int bcol1 = bcol0 + 16;
  const int koffB = (lane >> 4) * 16;
  const __bf16* pa_base  = A + arow * (size_t)K + koffA;
  const __bf16* pb0_base = W + (size_t)bcol0 * K + koffB;
  const __bf16* pb1_base = W + (size_t)bcol1 * K + koffB;

  v8f acc0 = {}, acc1 = {};
  for (int k0 = 0; k0 < K; k0 += 32) {
    FragBF a, b0, b1;
    const uint4* pa = (const uint4*)(pa_base + k0);
    a.u[0] = pa[0];          // K + 0..7
    a.u[1] = pa[1];          // K + 8..15
    const uint4* pb0 = (const uint4*)(pb0_base + k0);
    b0.u[0] = pb0[0]; b0.u[1] = pb0[1];
    const uint4* pb1 = (const uint4*)(pb1_base + k0);
    b1.u[0] = pb1[0]; b1.u[1] = pb1[1];
    acc0 = __builtin_amdgcn_wmma_f32_16x16x32_bf16(false, a.v, false, b0.v,
                                                   (short)0, acc0, false, false);
    acc1 = __builtin_amdgcn_wmma_f32_16x16x32_bf16(false, a.v, false, b1.v,
                                                   (short)0, acc1, false, false);
  }
  const float bv0 = bias[bcol0];
  const float bv1 = bias[bcol1];
  const size_t m0 = (size_t)tileM * 16 + (lane >> 4) * 8;
  #pragma unroll
  for (int r = 0; r < 8; ++r) {
    C[(m0 + r) * (size_t)N + bcol0] = acc0[r] + bv0;
    C[(m0 + r) * (size_t)N + bcol1] = acc1[r] + bv1;
  }
}

// ---------------------------------------------------------------------------
// 5) Recurrent scan (persistent): 16 workgroups x 16 batch rows, 8 waves.
//    Whh B-fragments preloaded into VGPRs for the whole scan; per step the
//    h A-fragments are read from LDS once and reused across owned N-tiles.
// ---------------------------------------------------------------------------
template <int H>
__global__ void gru_scan_kernel(const float* __restrict__ xp,   // (B*L, 3H) f32
                                const __bf16* __restrict__ Whh, // (3H, H) bf16
                                const float* __restrict__ bhh,  // (3H)
                                __bf16* __restrict__ outb,      // (B*L, H) bf16
                                float* __restrict__ outf,       // optional f32
                                int relu_out) {
  const int L   = 512;
  const int TH  = 3 * H;
  const int NT  = TH / 16;          // 16x16 N-tiles per step
  const int KS  = H / 32;           // WMMA k-steps
  const int TPW = (NT + 7) / 8;     // N-tiles per wave (H=128:3, 64:2, 32:1)
  extern __shared__ unsigned char smem[];
  float*  sb  = (float*)smem;            // TH      gate biases
  float*  sg  = sb + TH;                 // 16*TH   gate pre-activations (f32)
  float*  sh  = sg + 16 * TH;            // 16*H    hidden state f32
  __bf16* shb = (__bf16*)(sh + 16 * H);  // 16*H    hidden state bf16 (WMMA A)

  const int tid = threadIdx.x, lane = tid & 31, wave = tid >> 5;
  const int b0 = blockIdx.x * 16;

  for (int i = tid; i < TH; i += 256) sb[i] = bhh[i];
  for (int i = tid; i < 16 * H; i += 256) { sh[i] = 0.f; shb[i] = (__bf16)0.f; }

  const int arow  = lane & 15;
  const int koffA = (lane >> 4) * 16;
  const int koffB = (lane >> 4) * 16;

  // ---- hoist Whh B-fragments into registers for the whole scan ----------
  FragBF bfrag[TPW][KS];
  bool   own[TPW];
  int    ncol[TPW];
  #pragma unroll
  for (int i = 0; i < TPW; ++i) {
    const int nt = wave + 8 * i;
    own[i]  = (nt < NT);
    const int ntc = own[i] ? nt : (NT - 1);          // clamp: keep loads legal
    ncol[i] = ntc * 16 + (lane & 15);
    const __bf16* pb_base = Whh + (size_t)ncol[i] * H + koffB;
    #pragma unroll
    for (int k = 0; k < KS; ++k) {
      const uint4* pb = (const uint4*)(pb_base + 32 * k);
      bfrag[i][k].u[0] = pb[0];
      bfrag[i][k].u[1] = pb[1];
    }
  }
  __syncthreads();

  for (int t = 0; t < L; ++t) {
    // --- load h A-fragments once, reuse across all owned N-tiles ---------
    FragBF afrag[KS];
    #pragma unroll
    for (int k = 0; k < KS; ++k) {
      const uint4* pa = (const uint4*)(shb + arow * H + 32 * k + koffA);
      afrag[k].u[0] = pa[0];
      afrag[k].u[1] = pa[1];
    }
    // --- pure register WMMA: g = h @ Whh^T --------------------------------
    const int m0 = (lane >> 4) * 8;
    #pragma unroll
    for (int i = 0; i < TPW; ++i) {
      if (!own[i]) continue;                          // wave-uniform
      v8f acc = {};
      #pragma unroll
      for (int k = 0; k < KS; ++k)
        acc = __builtin_amdgcn_wmma_f32_16x16x32_bf16(
            false, afrag[k].v, false, bfrag[i][k].v, (short)0, acc, false, false);
      #pragma unroll
      for (int r = 0; r < 8; ++r) sg[(m0 + r) * TH + ncol[i]] = acc[r];
    }
    __syncthreads();
    // --- gate phase -------------------------------------------------------
    for (int idx = tid; idx < 16 * H; idx += 256) {   // exact multiple
      const int m = idx / H, j = idx - m * H;
      const size_t rowi = (size_t)(b0 + m) * L + t;
      const float* xr = xp + rowi * TH;
      if (t + 1 < L) __builtin_prefetch(xr + TH, 0, 0);  // next step's xp row
      const float gr = sg[m * TH + j]         + sb[j];
      const float gz = sg[m * TH + H + j]     + sb[H + j];
      const float gn = sg[m * TH + 2 * H + j] + sb[2 * H + j];
      const float r = 1.f / (1.f + __expf(-(xr[j]         + gr)));
      const float z = 1.f / (1.f + __expf(-(xr[H + j]     + gz)));
      const float n = tanhf(xr[2 * H + j] + r * gn);
      const float hp = sh[idx];
      const float hn = (1.f - z) * n + z * hp;
      sh[idx]  = hn;
      shb[idx] = (__bf16)hn;                           // recurrence uses raw h
      const float o = relu_out ? fmaxf(hn, 0.f) : hn;  // ReLU only on stack out
      outb[rowi * H + j] = (__bf16)o;
      if (outf) outf[rowi * H + j] = o;
    }
    __syncthreads();
  }
}

// ---------------------------------------------------------------------------
// 6) Loss: two-stage deterministic reduction.
// ---------------------------------------------------------------------------
__global__ void loss_partial_kernel(const float* __restrict__ M,
                                    const float* __restrict__ X,
                                    const float* __restrict__ G,
                                    const float* __restrict__ Dl,
                                    float* __restrict__ partial, int n) {
  float s1 = 0.f, s2 = 0.f, s3 = 0.f;
  for (int i = blockIdx.x * blockDim.x + threadIdx.x; i < n;
       i += gridDim.x * blockDim.x) {
    const float m = M[i];
    const float p = 1.f / (1.f + __expf(-Dl[i]));
    s1 += (1.f - m) * __logf(p + 1e-8f);
    const float d = X[i] - G[i];
    s2 += m * d * d;
    s3 += m;
  }
  __shared__ float sm[8][3];
  s1 = wave_sum(s1); s2 = wave_sum(s2); s3 = wave_sum(s3);
  if ((threadIdx.x & 31) == 0) {
    const int w = threadIdx.x >> 5;
    sm[w][0] = s1; sm[w][1] = s2; sm[w][2] = s3;
  }
  __syncthreads();
  if (threadIdx.x == 0) {
    float a = 0.f, b = 0.f, c = 0.f;
    for (int w = 0; w < 8; ++w) { a += sm[w][0]; b += sm[w][1]; c += sm[w][2]; }
    partial[blockIdx.x * 3 + 0] = a;
    partial[blockIdx.x * 3 + 1] = b;
    partial[blockIdx.x * 3 + 2] = c;
  }
}

__global__ void loss_final_kernel(const float* __restrict__ partial, int nb,
                                  float* __restrict__ out, float ntot) {
  float s1 = 0.f, s2 = 0.f, s3 = 0.f;
  for (int i = threadIdx.x; i < nb; i += blockDim.x) {
    s1 += partial[i * 3 + 0];
    s2 += partial[i * 3 + 1];
    s3 += partial[i * 3 + 2];
  }
  __shared__ float sm[8][3];
  s1 = wave_sum(s1); s2 = wave_sum(s2); s3 = wave_sum(s3);
  if ((threadIdx.x & 31) == 0) {
    const int w = threadIdx.x >> 5;
    sm[w][0] = s1; sm[w][1] = s2; sm[w][2] = s3;
  }
  __syncthreads();
  if (threadIdx.x == 0) {
    float a = 0.f, b = 0.f, c = 0.f;
    for (int w = 0; w < 8; ++w) { a += sm[w][0]; b += sm[w][1]; c += sm[w][2]; }
    const float g_loss = -a / ntot;          // -mean((1-M) log(D+eps))
    const float mse    = b / c;              // mean(M(X-G)^2)/mean(M)
    out[0] = g_loss + 100.f * mse;
  }
}

// ---------------------------------------------------------------------------
// Host orchestration
// ---------------------------------------------------------------------------
static void launch_layer(const __bf16* in, __bf16* outb, float* outf, float* xp,
                         const __bf16* wih, const __bf16* whh,
                         const float* bih, const float* bhh,
                         int din, int H, int relu, hipStream_t stream) {
  const size_t BL = 256 * 512;
  const int TH = 3 * H;
  dim3 gg(TH / 32, (unsigned)(BL / 64));
  gemm_bias_kernel<<<gg, 128, 0, stream>>>(in, wih, bih, xp, din, TH);
  const size_t sh = (size_t)(TH + 16 * TH + 16 * H) * 4 + (size_t)16 * H * 2;
  if (H == 128)
    gru_scan_kernel<128><<<16, 256, sh, stream>>>(xp, whh, bhh, outb, outf, relu);
  else if (H == 64)
    gru_scan_kernel<64><<<16, 256, sh, stream>>>(xp, whh, bhh, outb, outf, relu);
  else
    gru_scan_kernel<32><<<16, 256, sh, stream>>>(xp, whh, bhh, outb, outf, relu);
}

extern "C" void kernel_launch(void* const* d_in, const int* in_sizes, int n_in,
                              void* d_out, int out_size, void* d_ws, size_t ws_size,
                              hipStream_t stream) {
  (void)n_in; (void)out_size; (void)ws_size;
  const int B = 256, L = 512;
  const size_t BL = (size_t)B * L, BLD = BL * 32;

  const float* data  = (const float*)d_in[0];
  const float* obsm  = (const float*)d_in[1];
  const float* ratio = (const float*)d_in[2];
  const float* mrand = (const float*)d_in[3];
  const float* Z     = (const float*)d_in[4];
  const float* Htmp  = (const float*)d_in[5];

  unsigned char* ws = (unsigned char*)d_ws;
  size_t off = 0;
  auto alloc = [&](size_t bytes) -> void* {
    void* p = ws + off;
    off = (off + bytes + 255) & ~(size_t)255;
    return p;
  };
  float*  Mb   = (float*)alloc(BLD * 4);
  float*  Xb   = (float*)alloc(BLD * 4);
  float*  Gb   = (float*)alloc(BLD * 4);
  float*  Db   = (float*)alloc(BLD * 4);
  __bf16* actA = (__bf16*)alloc(BL * 128 * 2);
  __bf16* actB = (__bf16*)alloc(BL * 128 * 2);
  float*  xp   = (float*)alloc(BL * 384 * 4);
  float*  part = (float*)alloc(1024 * 3 * 4);

  // bf16 weight copies: layer li in [0,12): gen 0..5, disc 6..11;
  // d_in[6 + li*4 + {0:Wih,1:Whh,2:bih,3:bhh}]
  __bf16* wbf[12][2];
  for (int li = 0; li < 12; ++li) {
    for (int c = 0; c < 2; ++c) {
      const int ii = 6 + li * 4 + c;
      const int n = in_sizes[ii];
      wbf[li][c] = (__bf16*)alloc((size_t)n * 2);
      f32_to_bf16_kernel<<<(n + 1023) / 1024, 256, 0, stream>>>(
          (const float*)d_in[ii], wbf[li][c], n);
    }
  }

  // 1) top-k mask (64KB dynamic LDS per workgroup; WGP has 320KB)
  hipFuncSetAttribute((const void*)mask_topk_kernel,
                      hipFuncAttributeMaxDynamicSharedMemorySize, 16384 * 4);
  mask_topk_kernel<<<B, 1024, 16384 * 4, stream>>>(obsm, mrand, ratio, Mb);

  // 2) generator input: X = M*data + (1-M)*Z ; act = cat([X, M]) (bf16)
  const int nb = (int)((BLD + 255) / 256);
  build_gen_input_kernel<<<nb, 256, 0, stream>>>(data, Z, Mb, Xb, actA, (int)BLD);

  static const int DINS[6] = {64, 128, 128, 64, 64, 32};
  static const int HS[6]   = {128, 128, 64, 64, 32, 32};

  auto run_net = [&](int base_li, float* final_f32) {
    __bf16* cin = actA; __bf16* cout = actB;
    for (int l = 0; l < 6; ++l) {
      const int li = base_li + l;
      const int relu = (l == 1 || l == 3) ? 1 : 0;  // ReLU after stacks 1 and 2
      float* f32o = (l == 5) ? final_f32 : nullptr;
      const float* bih = (const float*)d_in[6 + li * 4 + 2];
      const float* bhh = (const float*)d_in[6 + li * 4 + 3];
      launch_layer(cin, cout, f32o, xp, wbf[li][0], wbf[li][1], bih, bhh,
                   DINS[l], HS[l], relu, stream);
      __bf16* t = cin; cin = cout; cout = t;
    }
  };

  // 3) generator (G_sample -> Gb)
  run_net(0, Gb);

  // 4) discriminator input: cat([X_hat, M*H_temp])
  build_disc_input_kernel<<<nb, 256, 0, stream>>>(Xb, Gb, Mb, Htmp, actA, (int)BLD);

  // 5) discriminator (pre-sigmoid logits -> Db)
  run_net(6, Db);

  // 6) losses
  loss_partial_kernel<<<1024, 256, 0, stream>>>(Mb, Xb, Gb, Db, part, (int)BLD);
  loss_final_kernel<<<1, 256, 0, stream>>>(part, 1024, (float*)d_out, (float)BLD);
}
